// Int8Linear_69956427317869
// MI455X (gfx1250) — matrix operand order repaired and checked by reading, not compiled, not run on hardware
//
#include <hip/hip_runtime.h>
#include <hip/hip_bf16.h>

typedef _Float16 v16h __attribute__((ext_vector_type(16)));
typedef _Float16 v8h  __attribute__((ext_vector_type(8)));
typedef float    v8f  __attribute__((ext_vector_type(8)));
typedef float    v4f  __attribute__((ext_vector_type(4)));
typedef int      v4i  __attribute__((ext_vector_type(4)));

#define M_TOT 2048
#define N_TOT 11008
#define K_TOT 4096

#define BM 128
#define BN 128
#define BK 32
#define LDSS 40   // 32 halfs + 8 pad (80B row stride: 16B aligned, bank-spread)

__global__ __launch_bounds__(256)
void int8linear_wmma_kernel(const float* __restrict__ x,
                            const int*   __restrict__ w,
                            const float* __restrict__ scales,
                            const float* __restrict__ bias,
                            float* __restrict__ out)
{
    // double-buffered tiles: one barrier per K-step
    __shared__ _Float16 ldsA[2][BM * LDSS];   // x tile  [row m][k], f16
    __shared__ _Float16 ldsB[2][BN * LDSS];   // w tile  [row n][k], f16

    const int tid  = threadIdx.x;
    const int lane = tid & 31;
    const int wave = tid >> 5;
    const int wm   = wave >> 2;       // 0..1 : 64-row slab in M
    const int wn   = wave & 3;        // 0..3 : 32-col slab in N

    const int m0 = blockIdx.y * BM;
    const int n0 = blockIdx.x * BN;

    // staging: 256 threads cover 128 rows x 32 k; thread -> (row, 16-k half)
    const int srow = tid >> 1;          // 0..127
    const int scol = (tid & 1) << 4;    // 0 or 16
    const int soff = srow * LDSS + scol;

    const float* xg = x + (size_t)(m0 + srow) * K_TOT + scol;
    const int*   wg = w + (size_t)(n0 + srow) * K_TOT + scol;

    v8f acc[4][2];
#pragma unroll
    for (int i = 0; i < 4; ++i)
#pragma unroll
        for (int j = 0; j < 2; ++j)
            acc[i][j] = (v8f){};

    // register staging buffers (16 floats / 16 ints per thread)
    v4f xr[4];
    v4i wr[4];

    const int ar = lane & 15;   // row/col within 16
    const int ah = lane >> 4;   // half-wave selector
    const int nk = K_TOT / BK;  // 128 k-steps

    // ---- prologue: tile 0 -> LDS buf0; tile 1 -> regs ----
#pragma unroll
    for (int j = 0; j < 4; ++j) {
        xr[j] = *(const v4f*)(xg + j * 4);
        wr[j] = *(const v4i*)(wg + j * 4);
    }
    {
        v8h h0{}, h1{}, g0{}, g1{};
#pragma unroll
        for (int e = 0; e < 4; ++e) {
            h0[e]     = (_Float16)xr[0][e];
            h0[4 + e] = (_Float16)xr[1][e];
            h1[e]     = (_Float16)xr[2][e];
            h1[4 + e] = (_Float16)xr[3][e];
            g0[e]     = (_Float16)(float)wr[0][e];
            g0[4 + e] = (_Float16)(float)wr[1][e];
            g1[e]     = (_Float16)(float)wr[2][e];
            g1[4 + e] = (_Float16)(float)wr[3][e];
        }
        *(v8h*)(&ldsA[0][soff])     = h0;
        *(v8h*)(&ldsA[0][soff + 8]) = h1;
        *(v8h*)(&ldsB[0][soff])     = g0;
        *(v8h*)(&ldsB[0][soff + 8]) = g1;
    }
#pragma unroll
    for (int j = 0; j < 4; ++j) {
        xr[j] = *(const v4f*)(xg + BK + j * 4);
        wr[j] = *(const v4i*)(wg + BK + j * 4);
    }
    __syncthreads();

    // ---- main loop: invariant at top of step ks:
    //      buf[ks&1] holds tile ks; regs hold tile ks+1 (if any) ----
    for (int ks = 0; ks < nk; ++ks) {
        const int cur = ks & 1;
        const _Float16* bufA = ldsA[cur];
        const _Float16* bufB = ldsB[cur];

        // fragment loads first (before the loadcnt-gated staging stores)
        v16h afrag[4];
#pragma unroll
        for (int mt = 0; mt < 4; ++mt) {
            const _Float16* ap = &bufA[(wm * 64 + mt * 16 + ar) * LDSS + ah * 8];
            v8h a0 = *(const v8h*)ap;
            v8h a1 = *(const v8h*)(ap + 16);
            afrag[mt] = __builtin_shufflevector(a0, a1,
                0, 1, 2, 3, 4, 5, 6, 7, 8, 9, 10, 11, 12, 13, 14, 15);
        }
        v16h bfrag[2];
#pragma unroll
        for (int nt = 0; nt < 2; ++nt) {
            const _Float16* bp = &bufB[(wn * 32 + nt * 16 + ar) * LDSS + ah * 8];
            v8h b0 = *(const v8h*)bp;
            v8h b1 = *(const v8h*)(bp + 16);
            bfrag[nt] = __builtin_shufflevector(b0, b1,
                0, 1, 2, 3, 4, 5, 6, 7, 8, 9, 10, 11, 12, 13, 14, 15);
        }

        // stage tile ks+1 (regs) into the other buffer, overlapped with WMMA
        if (ks + 1 < nk) {
            _Float16* nA = ldsA[cur ^ 1];
            _Float16* nB = ldsB[cur ^ 1];
            v8h h0{}, h1{}, g0{}, g1{};
#pragma unroll
            for (int e = 0; e < 4; ++e) {
                h0[e]     = (_Float16)xr[0][e];
                h0[4 + e] = (_Float16)xr[1][e];
                h1[e]     = (_Float16)xr[2][e];
                h1[4 + e] = (_Float16)xr[3][e];
                g0[e]     = (_Float16)(float)wr[0][e];
                g0[4 + e] = (_Float16)(float)wr[1][e];
                g1[e]     = (_Float16)(float)wr[2][e];
                g1[4 + e] = (_Float16)(float)wr[3][e];
            }
            *(v8h*)(&nA[soff])     = h0;
            *(v8h*)(&nA[soff + 8]) = h1;
            *(v8h*)(&nB[soff])     = g0;
            *(v8h*)(&nB[soff + 8]) = g1;

            // global loads for tile ks+2 issue now, retire next iteration
            if (ks + 2 < nk) {
                const float* xn  = xg + (size_t)(ks + 2) * BK;
                const int*   wg2 = wg + (size_t)(ks + 2) * BK;
#pragma unroll
                for (int j = 0; j < 4; ++j) {
                    xr[j] = *(const v4f*)(xn + j * 4);
                    wr[j] = *(const v4i*)(wg2 + j * 4);
                }
                if (ks + 3 < nk) {          // warm L2 ahead of the loads
                    __builtin_prefetch(xg + (size_t)(ks + 3) * BK, 0, 1);
                    __builtin_prefetch(wg + (size_t)(ks + 3) * BK, 0, 1);
                }
            }
        }

        // 8 WMMAs; VALU/DS/VMEM above co-execute under the XDL shadow
#pragma unroll
        for (int mt = 0; mt < 4; ++mt)
#pragma unroll
            for (int nt = 0; nt < 2; ++nt)
                acc[mt][nt] = __builtin_amdgcn_wmma_f32_16x16x32_f16(
                    false, afrag[mt], false, bfrag[nt],
                    (short)0, acc[mt][nt], false, false);

        __syncthreads();    // single barrier per K-step
    }

    // epilogue: per-output-channel scale + bias, fp32
#pragma unroll
    for (int nt = 0; nt < 2; ++nt) {
        const int o = n0 + wn * 32 + nt * 16 + ar;
        const float s  = scales[o];
        const float bz = bias[o];
#pragma unroll
        for (int mt = 0; mt < 4; ++mt) {
#pragma unroll
            for (int r = 0; r < 8; ++r) {
                const int m = m0 + wm * 64 + mt * 16 + r + (ah << 3);
                out[(size_t)m * N_TOT + o] = acc[mt][nt][r] * s + bz;
            }
        }
    }
}

extern "C" void kernel_launch(void* const* d_in, const int* in_sizes, int n_in,
                              void* d_out, int out_size, void* d_ws, size_t ws_size,
                              hipStream_t stream) {
    (void)in_sizes; (void)n_in; (void)out_size; (void)d_ws; (void)ws_size;
    const float* x      = (const float*)d_in[0];
    const int*   w      = (const int*)d_in[1];
    const float* scales = (const float*)d_in[2];
    const float* bias   = (const float*)d_in[3];
    float* out = (float*)d_out;

    dim3 grid(N_TOT / BN, M_TOT / BM);   // (86, 16)
    dim3 block(256);
    int8linear_wmma_kernel<<<grid, block, 0, stream>>>(x, w, scales, bias, out);
}
